// EGNN_11587821765357
// MI455X (gfx1250) — matrix operand order repaired and checked by reading, not compiled, not run on hardware
//
#include <hip/hip_runtime.h>
#include <hip/hip_bf16.h>

#define N_NODES   50000
#define N_EDGES   800000
#define N_GRAPHS  32
#define NODE_DIM  128
#define EDGE_DIM  128
#define HIDDEN    256
#define N_LAYERS  3

typedef __attribute__((ext_vector_type(16))) __bf16 v16bf;
typedef __attribute__((ext_vector_type(8)))  float  v8f;

union Frag {
    v16bf v;
    uint4 u[2];
};

// ---- optional CDNA5 async global->LDS path (guarded; falls back cleanly) ----
#if defined(__has_builtin)
#if __has_builtin(__builtin_amdgcn_global_load_async_to_lds_b128)
#define HAVE_ASYNC_LDS 1
#endif
#endif
#ifndef HAVE_ASYNC_LDS
#define HAVE_ASYNC_LDS 0
#endif

#if HAVE_ASYNC_LDS
// exact parameter type per hipcc diagnostic:
//   '__attribute__((__vector_size__(4 * sizeof(int)))) int __device__ *'
typedef int v4i_vs __attribute__((vector_size(4 * sizeof(int))));
typedef __attribute__((address_space(1))) v4i_vs* g_v4i_p;   // prints as __device__
typedef __attribute__((address_space(3))) v4i_vs* l_v4i_p;   // LDS
__device__ __forceinline__ void async_cp32(const unsigned short* gp, unsigned short* lp) {
    __builtin_amdgcn_global_load_async_to_lds_b128((g_v4i_p)gp, (l_v4i_p)lp, 0, 0);
    __builtin_amdgcn_global_load_async_to_lds_b128((g_v4i_p)(gp + 8), (l_v4i_p)(lp + 8), 0, 0);
}
__device__ __forceinline__ void async_wait0() {
#if __has_builtin(__builtin_amdgcn_s_wait_asynccnt)
    __builtin_amdgcn_s_wait_asynccnt(0);
#else
    asm volatile("s_wait_asynccnt 0x0" ::: "memory");
#endif
}
#else
__device__ __forceinline__ void async_cp32(const unsigned short* gp, unsigned short* lp) {
    ((uint4*)lp)[0] = ((const uint4*)gp)[0];
    ((uint4*)lp)[1] = ((const uint4*)gp)[1];
}
__device__ __forceinline__ void async_wait0() {}
#endif

__device__ __forceinline__ unsigned short f2bf(float f) {
    unsigned int u = __float_as_uint(f);
    unsigned int r = u + 0x7FFFu + ((u >> 16) & 1u);
    return (unsigned short)(r >> 16);
}

__device__ __forceinline__ float swishf(float x) {
    return x / (1.0f + __expf(-x));
}

// ---------------------------------------------------------------------------
// Weight packing: row-major fp32 W[K][N] -> per-fragment bf16 WMMA B layout.
// Fragment (ntile, kstep): 32 lanes x 16 bf16, lane l holds column
// n = ntile*16 + (l&15), elements e cover K = kstep*32 + ((l&16)?16:0) + e.
// ---------------------------------------------------------------------------
__global__ __launch_bounds__(256) void pack_w_kernel(
    const float* __restrict__ W, int ldw, int ksteps, int ntiles,
    unsigned short* __restrict__ out)
{
    int tid = blockIdx.x * blockDim.x + threadIdx.x;
    int total = ntiles * ksteps * 512;
    if (tid >= total) return;
    int frag = tid >> 9;
    int l    = (tid >> 4) & 31;
    int e    = tid & 15;
    int ntile = frag / ksteps;
    int ks    = frag % ksteps;
    int K = ks * 32 + ((l & 16) ? 16 : 0) + e;
    int n = ntile * 16 + (l & 15);
    out[tid] = f2bf(W[(size_t)K * ldw + n]);
}

// ---------------------------------------------------------------------------
// Edge geometry: d2[e] = |pos[dst]-pos[src] + shift @ lattice[batch[src]]|^2
// ---------------------------------------------------------------------------
__global__ __launch_bounds__(256) void edge_geom_kernel(
    const float* __restrict__ pos, const float* __restrict__ shift,
    const float* __restrict__ lat, const int* __restrict__ eidx,
    const int* __restrict__ batch, float* __restrict__ d2o)
{
    int e = blockIdx.x * blockDim.x + threadIdx.x;
    if (e >= N_EDGES) return;
    int s = eidx[e];
    int d = eidx[N_EDGES + e];
    int b = batch[s];
    const float* L = lat + (size_t)b * 9;
    float s0 = shift[e * 3 + 0], s1 = shift[e * 3 + 1], s2 = shift[e * 3 + 2];
    float acc = 0.f;
#pragma unroll
    for (int j = 0; j < 3; ++j) {
        float v = pos[(size_t)d * 3 + j] - pos[(size_t)s * 3 + j]
                + s0 * L[j] + s1 * L[3 + j] + s2 * L[6 + j];
        acc += v * v;
    }
    d2o[e] = acc;
}

// ---------------------------------------------------------------------------
// Embedding gather: x = embed[atomic_numbers], fp32 + bf16 copies
// ---------------------------------------------------------------------------
__global__ __launch_bounds__(256) void embed_kernel(
    const float* __restrict__ embed, const int* __restrict__ z,
    float* __restrict__ xf, unsigned short* __restrict__ xb)
{
    int tid = blockIdx.x * blockDim.x + threadIdx.x;
    if (tid >= N_NODES * NODE_DIM) return;
    int n = tid >> 7;
    int f = tid & 127;
    float v = embed[(size_t)z[n] * NODE_DIM + f];
    xf[tid] = v;
    xb[tid] = f2bf(v);
}

// ---------------------------------------------------------------------------
// WMMA fragment helpers. LDS activations use padded row stride LDW (bf16).
// A k order used here (contiguous, B packed alike): lane&15 = row; lanes<16 hold K 0-15,
// lanes>=16 hold K 16-31 -> two 16B ds loads per fragment.
// ---------------------------------------------------------------------------
#define LDW 264   // 528 B = 33*16 B : 16B-aligned rows, conflict-spread banks

template <int KS>
__device__ __forceinline__ void load_aK(const unsigned short* __restrict__ s_act,
                                        int lane, Frag (&a)[KS])
{
    const int mrow  = lane & 15;
    const int abase = (lane & 16) ? 16 : 0;
    const unsigned short* arow = s_act + mrow * LDW + abase;
#pragma unroll
    for (int ks = 0; ks < KS; ++ks) {
        a[ks].u[0] = *(const uint4*)(arow);
        a[ks].u[1] = *(const uint4*)(arow + 8);
        arow += 32;
    }
}

template <int KS>
__device__ __forceinline__ void load_bK(const unsigned short* __restrict__ wpack,
                                        int ntile, int lane, Frag (&b)[KS])
{
    const uint4* bp = (const uint4*)wpack + (size_t)ntile * KS * 64 + lane * 2;
#pragma unroll
    for (int ks = 0; ks < KS; ++ks) {
        b[ks].u[0] = bp[0];
        b[ks].u[1] = bp[1];
        bp += 64;
    }
}

// stream A from LDS against register-cached B fragments
template <int KS>
__device__ __forceinline__ v8f wmmaK_ab(const unsigned short* __restrict__ s_act,
                                        int lane, const Frag (&b)[KS])
{
    const int mrow  = lane & 15;
    const int abase = (lane & 16) ? 16 : 0;
    const unsigned short* arow = s_act + mrow * LDW + abase;
    v8f acc = {};
#pragma unroll
    for (int ks = 0; ks < KS; ++ks) {
        Frag a;
        a.u[0] = *(const uint4*)(arow);
        a.u[1] = *(const uint4*)(arow + 8);
        arow += 32;
        acc = __builtin_amdgcn_wmma_f32_16x16x32_bf16(
            false, a.v, false, b[ks].v, (short)0, acc, false, false);
    }
    return acc;
}

// stream B from global against register-cached A fragments
template <int KS>
__device__ __forceinline__ v8f wmmaK_ba(const Frag (&a)[KS],
                                        const unsigned short* __restrict__ wpack,
                                        int ntile, int lane)
{
    const uint4* bp = (const uint4*)wpack + (size_t)ntile * KS * 64 + lane * 2;
    v8f acc = {};
#pragma unroll
    for (int ks = 0; ks < KS; ++ks) {
        Frag b;
        b.u[0] = bp[0];
        b.u[1] = bp[1];
        bp += 64;
        acc = __builtin_amdgcn_wmma_f32_16x16x32_bf16(
            false, a[ks].v, false, b.v, (short)0, acc, false, false);
    }
    return acc;
}

// ---------------------------------------------------------------------------
// Edge (message) MLP for one layer + atomic scatter into agg.
// One block = 32 edges (two 16-row M tiles). 8 waves.
// GEMM1: concat(x[dst],x[src]) [32,256] x [256,256]; d2 enters as rank-1
// epilogue term (row 256 of edge_w1). GEMM2: [32,256] x [256,128].
// B fragments are register-cached and reused across both M tiles.
// ---------------------------------------------------------------------------
__global__ __launch_bounds__(256) void edge_mlp_kernel(
    const unsigned short* __restrict__ xbf, const float* __restrict__ d2,
    const int* __restrict__ eidx,
    const unsigned short* __restrict__ w1p, const float* __restrict__ w1c,
    const float* __restrict__ b1,
    const unsigned short* __restrict__ w2p, const float* __restrict__ b2,
    float* __restrict__ agg)
{
    __shared__ unsigned short s_in[32 * LDW];
    __shared__ unsigned short s_hid[32 * LDW];
    __shared__ int   s_dst[32];
    __shared__ float s_d2[32];

    const int tid  = threadIdx.x;
    const int lane = tid & 31;
    const int wave = tid >> 5;
    const int e0   = blockIdx.x * 32;

    if (tid < 32) {
        s_dst[tid] = eidx[N_EDGES + e0 + tid];
        s_d2[tid]  = d2[e0 + tid];
    }
    // stage concat(x[dst], x[src]) as 32 x 256 bf16 (async -> LDS if available)
#pragma unroll
    for (int q = 0; q < 2; ++q) {
        int c = tid + q * 256;
        int m = c >> 4;            // edge row 0..31
        int bblk = c & 15;         // 16-feature block
        int e = e0 + m;
        int node = (bblk < 8) ? eidx[N_EDGES + e] : eidx[e];
        int f0 = (bblk & 7) * 16;
        async_cp32(xbf + (size_t)node * NODE_DIM + f0,
                   s_in + m * LDW + bblk * 16);
    }
    async_wait0();
    __syncthreads();

    Frag bfr[8];

    // GEMM1: each wave owns ntiles {wave, wave+8}; cache B, stream A per M tile
#pragma unroll
    for (int t = 0; t < 2; ++t) {
        int ntile = wave + t * 8;
        load_bK<8>(w1p, ntile, lane, bfr);
        int n = ntile * 16 + (lane & 15);
        float bias = b1[n];
        float wc   = w1c[n];
        int mh = (lane & 16) ? 8 : 0;
#pragma unroll
        for (int mt = 0; mt < 2; ++mt) {
            v8f acc = wmmaK_ab<8>(s_in + mt * 16 * LDW, lane, bfr);
#pragma unroll
            for (int r = 0; r < 8; ++r) {
                int m = mt * 16 + r + mh;
                float v = swishf(acc[r] + bias + s_d2[m] * wc);
                s_hid[m * LDW + n] = f2bf(v);
            }
        }
    }
    __syncthreads();

    // GEMM2: wave owns one ntile of 128 outputs; swish then scatter-add
    {
        int ntile = wave;
        load_bK<8>(w2p, ntile, lane, bfr);
        int n = ntile * 16 + (lane & 15);
        float bias = b2[n];
        int mh = (lane & 16) ? 8 : 0;
#pragma unroll
        for (int mt = 0; mt < 2; ++mt) {
            v8f acc = wmmaK_ab<8>(s_hid + mt * 16 * LDW, lane, bfr);
#pragma unroll
            for (int r = 0; r < 8; ++r) {
                int m = mt * 16 + r + mh;
                float v = swishf(acc[r] + bias);
                unsafeAtomicAdd(&agg[(size_t)s_dst[m] * EDGE_DIM + n], v);
            }
        }
    }
}

// ---------------------------------------------------------------------------
// Node update MLP for one layer: concat(x, agg) -> 256 -> 128 (+residual).
// One block = 16 nodes; A fragments register-cached across N tiles.
// ---------------------------------------------------------------------------
__global__ __launch_bounds__(256) void node_mlp_kernel(
    unsigned short* __restrict__ xbf, const float* __restrict__ agg,
    const unsigned short* __restrict__ w1p, const float* __restrict__ b1,
    const unsigned short* __restrict__ w2p, const float* __restrict__ b2,
    float* __restrict__ xf)
{
    __shared__ unsigned short s_in[16 * LDW];
    __shared__ unsigned short s_hid[16 * LDW];

    const int tid  = threadIdx.x;
    const int lane = tid & 31;
    const int wave = tid >> 5;
    const int n0   = blockIdx.x * 16;

    {   // stage concat(x, agg) as 16 x 256 bf16
        int m = tid >> 4;
        int bblk = tid & 15;
        int node = n0 + m;
        int f0 = (bblk & 7) * 16;
        unsigned short* dp = s_in + m * LDW + bblk * 16;
        if (bblk < 8) {
            const uint4* sp = (const uint4*)(xbf + (size_t)node * NODE_DIM + f0);
            ((uint4*)dp)[0] = sp[0];
            ((uint4*)dp)[1] = sp[1];
        } else {
            const float4* ap = (const float4*)(agg + (size_t)node * EDGE_DIM + f0);
#pragma unroll
            for (int q = 0; q < 4; ++q) {
                float4 v = ap[q];
                dp[q * 4 + 0] = f2bf(v.x);
                dp[q * 4 + 1] = f2bf(v.y);
                dp[q * 4 + 2] = f2bf(v.z);
                dp[q * 4 + 3] = f2bf(v.w);
            }
        }
    }
    __syncthreads();

    Frag afr[8];

    // GEMM1: [16,256] x [256,256], swish; A cached, B streamed
    load_aK<8>(s_in, lane, afr);
#pragma unroll
    for (int t = 0; t < 2; ++t) {
        int ntile = wave + t * 8;
        v8f acc = wmmaK_ba<8>(afr, w1p, ntile, lane);
        int n = ntile * 16 + (lane & 15);
        float bias = b1[n];
        int mh = (lane & 16) ? 8 : 0;
#pragma unroll
        for (int r = 0; r < 8; ++r) {
            int m = r + mh;
            s_hid[m * LDW + n] = f2bf(swishf(acc[r] + bias));
        }
    }
    __syncthreads();

    // GEMM2: [16,256] x [256,128], + bias + residual (no activation)
    {
        load_aK<8>(s_hid, lane, afr);
        v8f acc = wmmaK_ba<8>(afr, w2p, wave, lane);
        int n = wave * 16 + (lane & 15);
        float bias = b2[n];
        int mh = (lane & 16) ? 8 : 0;
#pragma unroll
        for (int r = 0; r < 8; ++r) {
            int m = r + mh;
            size_t idx = (size_t)(n0 + m) * NODE_DIM + n;
            float v = xf[idx] + acc[r] + bias;
            xf[idx]  = v;
            xbf[idx] = f2bf(v);
        }
    }
}

// ---------------------------------------------------------------------------
// Readout: h = swish(x @ out_w1 + b1); scalar = h @ out_w2 + b2;
// per-graph atomic segment-sum. One block = 16 nodes.
// ---------------------------------------------------------------------------
__global__ __launch_bounds__(256) void readout_kernel(
    const unsigned short* __restrict__ xbf,
    const unsigned short* __restrict__ w1p, const float* __restrict__ b1,
    const float* __restrict__ w2, const float* __restrict__ b2,
    const int* __restrict__ batch, float* __restrict__ out)
{
    __shared__ unsigned short s_in[16 * LDW];
    __shared__ float s_red[16];

    const int tid  = threadIdx.x;
    const int lane = tid & 31;
    const int wave = tid >> 5;
    const int n0   = blockIdx.x * 16;

    if (tid < 16) s_red[tid] = 0.f;
    if (tid < 128) {   // stage x: 16 x 128 bf16
        int m = tid >> 3;
        int bblk = tid & 7;
        int node = n0 + m;
        const uint4* sp = (const uint4*)(xbf + (size_t)node * NODE_DIM + bblk * 16);
        uint4* dp = (uint4*)(s_in + m * LDW + bblk * 16);
        dp[0] = sp[0];
        dp[1] = sp[1];
    }
    __syncthreads();

    // GEMM: [16,128] x [128,256]; fold w2 into epilogue, reduce over hidden
    Frag afr[4];
    load_aK<4>(s_in, lane, afr);
#pragma unroll
    for (int t = 0; t < 2; ++t) {
        int ntile = wave + t * 8;
        v8f acc = wmmaK_ba<4>(afr, w1p, ntile, lane);
        int n = ntile * 16 + (lane & 15);
        float bias = b1[n];
        float wn   = w2[n];
        int mh = (lane & 16) ? 8 : 0;
#pragma unroll
        for (int r = 0; r < 8; ++r) {
            float v = swishf(acc[r] + bias) * wn;
            atomicAdd(&s_red[r + mh], v);
        }
    }
    __syncthreads();

    if (tid < 16) {
        int node = n0 + tid;
        float h = s_red[tid] + b2[0];
        unsafeAtomicAdd(&out[batch[node]], h);
    }
}

// ---------------------------------------------------------------------------
// Host launcher
// ---------------------------------------------------------------------------
extern "C" void kernel_launch(void* const* d_in, const int* in_sizes, int n_in,
                              void* d_out, int out_size, void* d_ws, size_t ws_size,
                              hipStream_t stream)
{
    const float* positions   = (const float*)d_in[0];
    const float* edge_shift  = (const float*)d_in[1];
    const float* lattice     = (const float*)d_in[2];
    const int*   atomic_nums = (const int*)  d_in[3];
    const int*   edge_index  = (const int*)  d_in[4];
    const int*   batch       = (const int*)  d_in[5];
    const float* embed       = (const float*)d_in[6];
    const float* edge_w1     = (const float*)d_in[7];
    const float* edge_b1     = (const float*)d_in[8];
    const float* edge_w2     = (const float*)d_in[9];
    const float* edge_b2     = (const float*)d_in[10];
    const float* node_w1     = (const float*)d_in[11];
    const float* node_b1     = (const float*)d_in[12];
    const float* node_w2     = (const float*)d_in[13];
    const float* node_b2     = (const float*)d_in[14];
    const float* out_w1      = (const float*)d_in[15];
    const float* out_b1      = (const float*)d_in[16];
    const float* out_w2      = (const float*)d_in[17];
    const float* out_b2      = (const float*)d_in[18];
    float* out = (float*)d_out;

    // workspace layout
    char* ws = (char*)d_ws;
    size_t off = 0;
    auto walloc = [&](size_t bytes) -> void* {
        void* p = ws + off;
        off = (off + bytes + 255) & ~(size_t)255;
        return p;
    };
    float*          d2v = (float*)walloc((size_t)N_EDGES * 4);
    float*          xf  = (float*)walloc((size_t)N_NODES * NODE_DIM * 4);
    unsigned short* xb  = (unsigned short*)walloc((size_t)N_NODES * NODE_DIM * 2);
    float*          agg = (float*)walloc((size_t)N_NODES * EDGE_DIM * 4);
    unsigned short* ew1p = (unsigned short*)walloc((size_t)N_LAYERS * 65536 * 2);
    unsigned short* ew2p = (unsigned short*)walloc((size_t)N_LAYERS * 32768 * 2);
    unsigned short* nw1p = (unsigned short*)walloc((size_t)N_LAYERS * 65536 * 2);
    unsigned short* nw2p = (unsigned short*)walloc((size_t)N_LAYERS * 32768 * 2);
    unsigned short* ow1p = (unsigned short*)walloc((size_t)32768 * 2);

    (void)hipMemsetAsync(d_out, 0, (size_t)out_size * sizeof(float), stream);

    // pack weights into WMMA B-fragment layout (bf16)
    for (int l = 0; l < N_LAYERS; ++l) {
        pack_w_kernel<<<65536 / 256, 256, 0, stream>>>(
            edge_w1 + (size_t)l * 257 * 256, 256, 8, 16, ew1p + (size_t)l * 65536);
        pack_w_kernel<<<32768 / 256, 256, 0, stream>>>(
            edge_w2 + (size_t)l * 256 * 128, 128, 8, 8, ew2p + (size_t)l * 32768);
        pack_w_kernel<<<65536 / 256, 256, 0, stream>>>(
            node_w1 + (size_t)l * 256 * 256, 256, 8, 16, nw1p + (size_t)l * 65536);
        pack_w_kernel<<<32768 / 256, 256, 0, stream>>>(
            node_w2 + (size_t)l * 256 * 128, 128, 8, 8, nw2p + (size_t)l * 32768);
    }
    pack_w_kernel<<<32768 / 256, 256, 0, stream>>>(out_w1, 256, 4, 16, ow1p);

    embed_kernel<<<(N_NODES * NODE_DIM) / 256, 256, 0, stream>>>(
        embed, atomic_nums, xf, xb);
    edge_geom_kernel<<<(N_EDGES + 255) / 256, 256, 0, stream>>>(
        positions, edge_shift, lattice, edge_index, batch, d2v);

    for (int l = 0; l < N_LAYERS; ++l) {
        (void)hipMemsetAsync(agg, 0, (size_t)N_NODES * EDGE_DIM * 4, stream);
        edge_mlp_kernel<<<N_EDGES / 32, 256, 0, stream>>>(
            xb, d2v, edge_index,
            ew1p + (size_t)l * 65536,
            edge_w1 + (size_t)l * 257 * 256 + (size_t)256 * 256,   // d2 row
            edge_b1 + (size_t)l * 256,
            ew2p + (size_t)l * 32768,
            edge_b2 + (size_t)l * 128,
            agg);
        node_mlp_kernel<<<N_NODES / 16, 256, 0, stream>>>(
            xb, agg,
            nw1p + (size_t)l * 65536, node_b1 + (size_t)l * 256,
            nw2p + (size_t)l * 32768, node_b2 + (size_t)l * 128,
            xf);
    }

    readout_kernel<<<N_NODES / 16, 256, 0, stream>>>(
        xb, ow1p, out_b1, out_w2, out_b2, batch, out);
}
